// MultiheadAttention_33732673142980
// MI455X (gfx1250) — matrix operand order repaired and checked by reading, not compiled, not run on hardware
//
#include <hip/hip_runtime.h>

// Problem constants (match reference)
#define BB   4
#define TT   2048
#define DIN  1024
#define DEMB 1024
#define HH   16
#define HD   64
#define MM   (BB*TT)     // 8192 rows
#define EQKV (3*DEMB)    // 3072 qkv channels

#if defined(__has_builtin)
#if __has_builtin(__builtin_amdgcn_global_load_async_to_lds_b128)
#define ASYNC_LDS 1
#endif
#endif
#ifndef ASYNC_LDS
#define ASYNC_LDS 0
#endif

typedef __attribute__((ext_vector_type(16))) __bf16 v16bf;
typedef __attribute__((ext_vector_type(8)))  float  v8f;
typedef __attribute__((__vector_size__(4 * sizeof(int)))) int v4i_g;   // builtin's pointee type

union Frag16 {
  v16bf v;
  unsigned short u[16];
  uint4 q[2];
};

__device__ __forceinline__ unsigned short f2bf(float f) {
  unsigned int u = __float_as_uint(f);
  u += 0x7FFFu + ((u >> 16) & 1u);   // round-to-nearest-even
  return (unsigned short)(u >> 16);
}

__device__ __forceinline__ v8f wmma_bf16(v16bf a, v16bf b, v8f c) {
  // D = A(16x32) * B(32x16) + C, f32 accumulate
  return __builtin_amdgcn_wmma_f32_16x16x32_bf16(false, a, false, b, (short)0, c, false, false);
}

// A fragment: 16x32 bf16 tile from row-major [row][col] matrix.
// ISA layout: lanes 0-15 hold M=0-15 K={0..7,16..23}; lanes 16-31 K={8..15,24..31}.
__device__ __forceinline__ v16bf load_a(const unsigned short* base, int m0, int k0, int ld) {
  const int lane = threadIdx.x & 31;
  const int half = lane >> 4;
  const unsigned short* p = base + (size_t)(m0 + (lane & 15)) * ld + k0 + half * 8;
  Frag16 f;
  f.q[0] = *(const uint4*)p;          // K = k0 + half*8 .. +7
  f.q[1] = *(const uint4*)(p + 16);   // K = k0 + 16 + half*8 .. +7
  return f.v;
}

// B fragment (32x16) with B[k][n] = W[(n0+n)*ld + k0+k]   (row-major [n][k] source)
// ISA layout: lane holds N=lane%16, K = (lane/16)*8 + {0..7, 16..23} (matches load_a).
__device__ __forceinline__ v16bf load_bt(const unsigned short* w, int n0, int k0, int ld) {
  const int lane = threadIdx.x & 31;
  const unsigned short* p = w + (size_t)(n0 + (lane & 15)) * ld + k0 + (lane >> 4) * 8;
  Frag16 f;
  f.q[0] = *(const uint4*)p;
  f.q[1] = *(const uint4*)(p + 16);
  return f.v;
}

// 16-byte global -> LDS copy: async DMA (ASYNCcnt) on CDNA5, fallback via VGPRs.
__device__ __forceinline__ void cp16(unsigned short* l, const unsigned short* g) {
#if ASYNC_LDS
  __builtin_amdgcn_global_load_async_to_lds_b128(
      (__attribute__((address_space(1))) v4i_g*)(__UINTPTR_TYPE__)(size_t)g,
      (__attribute__((address_space(3))) v4i_g*)l, 0, 0);
#else
  *(uint4*)l = *(const uint4*)g;
#endif
}

// -------------------- f32 -> bf16 conversion --------------------
__global__ void cvt_bf16(const float* __restrict__ src, unsigned short* __restrict__ dst, int n) {
  int i = blockIdx.x * blockDim.x + threadIdx.x;
  int stride = gridDim.x * blockDim.x;
  for (; i < n; i += stride) dst[i] = f2bf(src[i]);
}

// -------------------- QKV projection GEMM --------------------
// qkv[m][e] = sum_k x[m][k] * w_qkv[e][k]    M=8192, N=3072, K=1024
// 32x64 output per wave (2 A-frags x 4 B-frags -> 8 WMMA per k-step).
// A wave's 64-column segment is one whole q/k/v head-segment (192 = 3*64):
//   q,k columns -> token-major qkv buffer; v columns -> transposed [B,H,HD,T] buffer.
__global__ __launch_bounds__(256) void qkv_gemm(const unsigned short* __restrict__ xb,
                                                const unsigned short* __restrict__ wb,
                                                unsigned short* __restrict__ qkv,
                                                unsigned short* __restrict__ vT) {
  const int wave = threadIdx.x >> 5;
  const int lane = threadIdx.x & 31;
  const int m0 = blockIdx.x * 32;
  const int n0 = blockIdx.y * 512 + wave * 64;

  v8f acc[2][4] = {};
  for (int k = 0; k < DIN; k += 32) {
    v16bf a0 = load_a(xb, m0,      k, DIN);
    v16bf a1 = load_a(xb, m0 + 16, k, DIN);
#pragma unroll
    for (int n = 0; n < 4; ++n) {
      v16bf b = load_bt(wb, n0 + n * 16, k, DIN);
      acc[0][n] = wmma_bf16(a0, b, acc[0][n]);
      acc[1][n] = wmma_bf16(a1, b, acc[1][n]);
    }
  }

  const int half = lane >> 4, col = lane & 15;
  if ((n0 % 192) < 128) {
    // q or k segment: store token-major [m][e]
#pragma unroll
    for (int mt = 0; mt < 2; ++mt)
#pragma unroll
      for (int n = 0; n < 4; ++n)
#pragma unroll
        for (int r = 0; r < 8; ++r)
          qkv[(size_t)(m0 + mt * 16 + r + 8 * half) * EQKV + n0 + n * 16 + col] =
              f2bf(acc[mt][n][r]);
  } else {
    // v segment: store transposed vT[b][h][d][t]
    const int h = n0 / 192;
#pragma unroll
    for (int mt = 0; mt < 2; ++mt)
#pragma unroll
      for (int n = 0; n < 4; ++n)
#pragma unroll
        for (int r = 0; r < 8; ++r) {
          const int tok = m0 + mt * 16 + r + 8 * half;
          const int d = n * 16 + col;
          const int bb = tok / TT, t = tok % TT;
          vT[((size_t)(bb * HH + h) * HD + d) * TT + t] = f2bf(acc[mt][n][r]);
        }
  }
}

// -------------------- Flash-style causal attention --------------------
// Block = 8 waves, all on the same (b,h), 8 consecutive 16-query tiles.
// K/V tiles staged cooperatively into double-buffered LDS (async-to-LDS on CDNA5),
// shared by all 8 waves. Fully-masked tiles are exact no-ops in the online softmax,
// so the key loop is block-uniform (EXEC all-ones at every WMMA).
__global__ __launch_bounds__(256) void attn_kernel(const unsigned short* __restrict__ qkv,
                                                   const unsigned short* __restrict__ vT,
                                                   unsigned short* __restrict__ attn_out) {
  __shared__ __align__(16) unsigned short k_lds[2][32 * 64];  // [key][d]
  __shared__ __align__(16) unsigned short v_lds[2][64 * 32];  // [d][key]
  __shared__ __align__(16) unsigned short p_lds[8][16 * 32];  // per-wave P staging

  const int tid  = threadIdx.x;
  const int lane = tid & 31;
  const int wave = tid >> 5;
  const int Tq = TT / 16;                     // 128 query tiles per (b,h)
  const int b  = blockIdx.x / (HH * (Tq / 8));
  const int r0 = blockIdx.x % (HH * (Tq / 8));
  const int h  = r0 / (Tq / 8);
  const int qg = r0 % (Tq / 8);
  const int qt = qg * 8 + wave;               // this wave's 16-query tile

  const size_t bbase = (size_t)b * TT * EQKV;
  const unsigned short* qbase = qkv + bbase + h * 192;        // Q channels
  const unsigned short* kbase = qkv + bbase + h * 192 + 64;   // K channels
  const unsigned short* vbase = vT + (size_t)(b * HH + h) * HD * TT;

  // Q tile as two A fragments covering head-dim 0-31 / 32-63
  const v16bf qa0 = load_a(qbase, qt * 16, 0,  EQKV);
  const v16bf qa1 = load_a(qbase, qt * 16, 32, EQKV);

  v8f o0 = {}, o1 = {}, o2 = {}, o3 = {};
  float mrow[8], lrow[8];
#pragma unroll
  for (int r = 0; r < 8; ++r) { mrow[r] = -__builtin_inff(); lrow[r] = 0.f; }

  const int half = lane >> 4;
  const int col  = lane & 15;
  const int qi_base = qt * 16;
  unsigned short* pl = p_lds[wave];

  // Cooperative K/V tile staging: K 32x64 (4KB) + V 64x32 (4KB), 16B per thread each.
  const int kkey = tid >> 3, kchunk = tid & 7;   // K: 32 rows x 8 chunks
  const int vd   = tid >> 2, vchunk = tid & 3;   // V: 64 rows x 4 chunks

  const int jmaxb = ((qg * 8 + 7) * 16 + 15) >> 5;   // block-max key tile (uniform)

  // prologue: stage tile 0 into buffer 0
  cp16(k_lds[0] + kkey * 64 + kchunk * 8, kbase + (size_t)kkey * EQKV + kchunk * 8);
  cp16(v_lds[0] + vd * 32 + vchunk * 8,   vbase + (size_t)vd * TT + vchunk * 8);

  for (int jj = 0; jj <= jmaxb; ++jj) {
    const int cur = jj & 1;
    const int key0 = jj * 32;

    if (jj < jmaxb) {
      const int nk0 = (jj + 1) * 32;
      cp16(k_lds[cur ^ 1] + kkey * 64 + kchunk * 8,
           kbase + (size_t)(nk0 + kkey) * EQKV + kchunk * 8);
      cp16(v_lds[cur ^ 1] + vd * 32 + vchunk * 8,
           vbase + (size_t)vd * TT + nk0 + vchunk * 8);
#if ASYNC_LDS
      asm volatile("s_wait_asynccnt 0x2" ::: "memory");  // current tile's DMA done
#endif
    } else {
#if ASYNC_LDS
      asm volatile("s_wait_asynccnt 0x0" ::: "memory");
#endif
    }
    __syncthreads();   // all waves' staging of buffer `cur` complete

    const unsigned short* kt = k_lds[cur];
    const unsigned short* vt = v_lds[cur];

    // S = Q * K^T for 32 keys: two 16x16 D tiles, head-dim 64 = 2 WMMA each (from LDS)
    v8f s0 = {}, s1 = {};
    s0 = wmma_bf16(qa0, load_bt(kt, 0,  0,  HD), s0);
    s0 = wmma_bf16(qa1, load_bt(kt, 0,  32, HD), s0);
    s1 = wmma_bf16(qa0, load_bt(kt, 16, 0,  HD), s1);
    s1 = wmma_bf16(qa1, load_bt(kt, 16, 32, HD), s1);

    // Online softmax update (rows striped: M = r + 8*half, N = col across 16 lanes)
#pragma unroll
    for (int r = 0; r < 8; ++r) {
      const int qi = qi_base + r + 8 * half;
      float e0 = s0[r] * 0.125f;          // 1/sqrt(64)
      float e1 = s1[r] * 0.125f;
      if (key0 + col      > qi) e0 = -__builtin_inff();
      if (key0 + 16 + col > qi) e1 = -__builtin_inff();
      float mx = fmaxf(e0, e1);
#pragma unroll
      for (int d = 8; d >= 1; d >>= 1) mx = fmaxf(mx, __shfl_xor(mx, d, 32));
      const float mn = fmaxf(mrow[r], mx);
      const float sc = __expf(mrow[r] - mn);
      const float p0 = __expf(e0 - mn);
      const float p1 = __expf(e1 - mn);
      float rs = p0 + p1;
#pragma unroll
      for (int d = 8; d >= 1; d >>= 1) rs += __shfl_xor(rs, d, 32);
      lrow[r] = lrow[r] * sc + rs;
      mrow[r] = mn;
      o0[r] *= sc; o1[r] *= sc; o2[r] *= sc; o3[r] *= sc;
      const int M = r + 8 * half;
      pl[M * 32 + col]      = f2bf(p0);
      pl[M * 32 + 16 + col] = f2bf(p1);
    }

    asm volatile("s_wait_dscnt 0x0" ::: "memory");   // P writes visible before A-frag read
    const v16bf pa = load_a(pl, 0, 0, 32);           // P as 16x32 A fragment (from LDS)

    // O += P * V  (4 head-dim column tiles), V tile is [d][key] in LDS
    o0 = wmma_bf16(pa, load_bt(vt, 0,  0, 32), o0);
    o1 = wmma_bf16(pa, load_bt(vt, 16, 0, 32), o1);
    o2 = wmma_bf16(pa, load_bt(vt, 32, 0, 32), o2);
    o3 = wmma_bf16(pa, load_bt(vt, 48, 0, 32), o3);

    __syncthreads();   // all reads of buffer `cur` done before it is restaged
  }

  // Normalize and store bf16 attention output [B*T][DEMB], head h at column h*64
#pragma unroll
  for (int r = 0; r < 8; ++r) {
    const int M = r + 8 * half;
    const float inv = 1.f / lrow[r];
    const size_t row = (size_t)(b * TT + qt * 16 + M) * DEMB + h * HD + col;
    attn_out[row + 0 ] = f2bf(o0[r] * inv);
    attn_out[row + 16] = f2bf(o1[r] * inv);
    attn_out[row + 32] = f2bf(o2[r] * inv);
    attn_out[row + 48] = f2bf(o3[r] * inv);
  }
}

// -------------------- Output projection GEMM --------------------
// out[m][o] = sum_e attn[m][e] * w_o[o][e] + b_o[o]    M=8192, N=1024, K=1024
__global__ __launch_bounds__(256) void o_proj(const unsigned short* __restrict__ attn,
                                              const unsigned short* __restrict__ wob,
                                              const float* __restrict__ b_o,
                                              float* __restrict__ out) {
  const int wave = threadIdx.x >> 5;
  const int lane = threadIdx.x & 31;
  const int m0 = blockIdx.x * 32;
  const int n0 = blockIdx.y * 512 + wave * 64;

  v8f acc[2][4] = {};
  for (int k = 0; k < DEMB; k += 32) {
    v16bf a0 = load_a(attn, m0,      k, DEMB);
    v16bf a1 = load_a(attn, m0 + 16, k, DEMB);
#pragma unroll
    for (int n = 0; n < 4; ++n) {
      v16bf b = load_bt(wob, n0 + n * 16, k, DEMB);
      acc[0][n] = wmma_bf16(a0, b, acc[0][n]);
      acc[1][n] = wmma_bf16(a1, b, acc[1][n]);
    }
  }

  const int half = lane >> 4, col = lane & 15;
#pragma unroll
  for (int n = 0; n < 4; ++n) {
    const float bias = b_o[n0 + n * 16 + col];
#pragma unroll
    for (int mt = 0; mt < 2; ++mt)
#pragma unroll
      for (int r = 0; r < 8; ++r)
        out[(size_t)(m0 + mt * 16 + r + 8 * half) * DEMB + n0 + n * 16 + col] =
            acc[mt][n][r] + bias;
  }
}

// -------------------- host launcher --------------------
extern "C" void kernel_launch(void* const* d_in, const int* in_sizes, int n_in,
                              void* d_out, int out_size, void* d_ws, size_t ws_size,
                              hipStream_t stream) {
  const float* x     = (const float*)d_in[0];   // [B,T,DIN]
  const float* w_qkv = (const float*)d_in[1];   // [3*DEMB, DIN]
  const float* w_o   = (const float*)d_in[2];   // [DEMB, DEMB]
  const float* b_o   = (const float*)d_in[3];   // [DEMB]
  float* out = (float*)d_out;                   // [B,T,DEMB] f32

  char* ws = (char*)d_ws;
  unsigned short* xb   = (unsigned short*)(ws);                              // 16 MB
  unsigned short* wqb  = (unsigned short*)(ws + (size_t)16 * 1024 * 1024);   //  6 MB
  unsigned short* wob  = (unsigned short*)(ws + (size_t)23 * 1024 * 1024);   //  2 MB
  unsigned short* qkvb = (unsigned short*)(ws + (size_t)26 * 1024 * 1024);   // 48 MB
  unsigned short* vT   = (unsigned short*)(ws + (size_t)75 * 1024 * 1024);   // 16 MB
  unsigned short* attn = (unsigned short*)(ws + (size_t)91 * 1024 * 1024);   // 16 MB

  cvt_bf16<<<4096, 256, 0, stream>>>(x,     xb,  MM * DIN);
  cvt_bf16<<<4096, 256, 0, stream>>>(w_qkv, wqb, EQKV * DIN);
  cvt_bf16<<<4096, 256, 0, stream>>>(w_o,   wob, DEMB * DEMB);

  qkv_gemm<<<dim3(MM / 32, EQKV / 512), 256, 0, stream>>>(xb, wqb, qkvb, vT);

  attn_kernel<<<dim3(BB * HH * (TT / 16) / 8), 256, 0, stream>>>(qkvb, vT, attn);

  o_proj<<<dim3(MM / 32, DEMB / 512), 256, 0, stream>>>(attn, wob, b_o, out);
}
